// SpatialAttention_40020505264464
// MI455X (gfx1250) — matrix operand order repaired and checked by reading, not compiled, not run on hardware
//
#include <hip/hip_runtime.h>
#include <hip/hip_bf16.h>
#include <math.h>

// ---------------------------------------------------------------------------
// SpatialAttention fused kernel for gfx1250 (MI455X).
//   B=512, C=512, S=7 (49 spatial positions, padded to N=64 for WMMA tiles).
// One workgroup (256 threads = 8 wave32) handles one batch:
//   P0: transpose x[b] -> LDS xT[n][c] (f16)            (B operand layout)
//   P1: q,k,v = W*x + b  via v_wmma_f32_16x16x32_f16    (A = f16 weights in L2)
//   P2: scores[c] = q[c](7x7) @ k[c](7x7)               (scalar, tiny)
//   P3: softmax over channels at each of 49 positions   (4-way split reduction)
//   P4: attn[c] = sm[c] @ v[c] -> LDS attnT[n][c]       (scalar, tiny)
//   P5: out = alpha*(Wr*attn + br) + x                  (WMMA + fused epilogue)
// ---------------------------------------------------------------------------

typedef __attribute__((ext_vector_type(16))) _Float16 v16h;
typedef __attribute__((ext_vector_type(8)))  _Float16 v8h;
typedef __attribute__((ext_vector_type(8)))  float    v8f;

#define CHN   512
#define SS    7
#define SP    49          // 7*7
#define NPAD  64          // spatial padded to 4 WMMA N-tiles
#define KSTEP 32          // K per v_wmma_f32_16x16x32_f16
#define PITCH_T 520       // halfs: row pitch of xT/attnT (64B-friendly, bank-skewed)
#define PITCH_C 52        // halfs: row pitch of per-channel q/k/v/scores

// LDS layout (bytes), phase-aliased to fit under 320KB/WGP:
//   [0,66560)        xT  (64*520*2)    ... later: scores f16 (512*52*2=53248)
//   [66560,119808)   q   (512*52*2)    ... later: attnT rows (64*520*2=66560
//   [119808,173056)  k   (512*52*2)         spans q+k region; softmax scratch
//   [173056,226304)  v   (512*52*2)         sits at 133120, past attnT's end)
#define OFF_XT   0
#define OFF_Q    66560
#define OFF_K    119808
#define OFF_V    173056
#define OFF_RED  133120   // 2*196 floats scratch, in dead tail of q+k region
#define LDS_BYTES 226304

// --- fragment loaders -------------------------------------------------------
// A (16-bit 16x32): lane l holds row m=l&15; 32-bit regs 0..3 = K kb..kb+7,
// regs 4..7 = K kb+16..kb+23, kb = (l>=16 ? 8 : 0).  -> two 16B loads.
__device__ __forceinline__ v16h load_a_frag(const _Float16* __restrict__ row,
                                            int kb) {
  v8h lo = *(const v8h*)(row + kb);
  v8h hi = *(const v8h*)(row + kb + 16);
  return __builtin_shufflevector(lo, hi, 0, 1, 2, 3, 4, 5, 6, 7,
                                 8, 9, 10, 11, 12, 13, 14, 15);
}
// B (16-bit 32x16): lane l holds col n=l&15; same K order as A:
// elems 0..7 = K kb/2..kb/2+7, elems 8..15 = K kb/2+16..kb/2+23 -> two 16B loads.
__device__ __forceinline__ v16h load_b_frag(const _Float16* __restrict__ row,
                                            int kb) {
  v8h lo = *(const v8h*)(row + (kb >> 1));
  v8h hi = *(const v8h*)(row + (kb >> 1) + 16);
  return __builtin_shufflevector(lo, hi, 0, 1, 2, 3, 4, 5, 6, 7,
                                 8, 9, 10, 11, 12, 13, 14, 15);
}

// f32 -> f16 weight prepack (runs once per launch; weights live in L2 after)
extern "C" __global__ void __launch_bounds__(256)
cvt_w_f16(const float* __restrict__ src, _Float16* __restrict__ dst, int n) {
  int i = blockIdx.x * blockDim.x + threadIdx.x;
  if (i < n) dst[i] = (_Float16)src[i];
}

extern "C" __global__ void __launch_bounds__(256, 1)
spatial_attn_fused(const float* __restrict__ x,
                   const _Float16* __restrict__ W16,  // [4][512][512] q,k,v,r
                   const float* __restrict__ bq, const float* __restrict__ bk,
                   const float* __restrict__ bv, const float* __restrict__ br,
                   const float* __restrict__ alpha,
                   float* __restrict__ out) {
  extern __shared__ __align__(16) char smem[];
  _Float16* xT = (_Float16*)(smem + OFF_XT);
  _Float16* qc = (_Float16*)(smem + OFF_Q);
  _Float16* kc = (_Float16*)(smem + OFF_K);
  _Float16* vc = (_Float16*)(smem + OFF_V);
  _Float16* sc = (_Float16*)(smem + OFF_XT);  // scores alias (xT dead)
  _Float16* aT = (_Float16*)(smem + OFF_Q);   // attnT alias (q+k dead)
  float*    redm = (float*)(smem + OFF_RED);  // softmax partial maxes [49][4]
  float*    reds = redm + 196;                // softmax partial sums  [49][4]

  const int b    = blockIdx.x;
  const int t    = threadIdx.x;
  const int lane = t & 31;
  const int wave = t >> 5;
  const int nl   = lane & 15;
  const int hi   = (lane >= 16) ? 1 : 0;
  const int kb_a = hi * 8;    // A-frag K sub-block
  const int kb_b = hi * 16;   // B-frag K sub-block
  const float* __restrict__ xb = x + (size_t)b * CHN * SP;

  // ---- P0: zero pad + transpose x[b] into xT[n][c] (f16) ----
  for (int i = t; i < NPAD * PITCH_T; i += 256) xT[i] = (_Float16)0.f;
  __syncthreads();
  for (int i = t; i < CHN * SP; i += 256) {
    int c = i / SP, n = i - c * SP;
    xT[n * PITCH_T + c] = (_Float16)xb[i];
  }
  __syncthreads();

  // ---- P1: q,k,v GEMMs (M=512 split 64/wave, N=64, K=512) ----
  const int mrow = wave * 64;
  for (int g = 0; g < 3; ++g) {
    const _Float16* __restrict__ W = W16 + (size_t)g * CHN * CHN;
    const float* __restrict__ bias = (g == 0) ? bq : (g == 1) ? bk : bv;
    _Float16* __restrict__ dst     = (g == 0) ? qc : (g == 1) ? kc : vc;

    v8f acc[4][4];
#pragma unroll
    for (int i = 0; i < 4; ++i)
#pragma unroll
      for (int j = 0; j < 4; ++j) acc[i][j] = (v8f){0, 0, 0, 0, 0, 0, 0, 0};

    for (int kk = 0; kk < CHN; kk += KSTEP) {
      v16h bf[4];
#pragma unroll
      for (int nt = 0; nt < 4; ++nt)
        bf[nt] = load_b_frag(xT + (nt * 16 + nl) * PITCH_T + kk, kb_b);
#pragma unroll
      for (int mt = 0; mt < 4; ++mt) {
        v16h af = load_a_frag(W + (size_t)(mrow + mt * 16 + nl) * CHN + kk, kb_a);
#pragma unroll
        for (int nt = 0; nt < 4; ++nt)
          acc[mt][nt] = __builtin_amdgcn_wmma_f32_16x16x32_f16(
              false, af, false, bf[nt], (short)0, acc[mt][nt], false, false);
      }
    }
    // bias + f16 store into channel-major LDS [c][49]
#pragma unroll
    for (int mt = 0; mt < 4; ++mt) {
      const int mb = mrow + mt * 16 + kb_a;     // reg r -> m = mb + r
      v8f bvv = *(const v8f*)(bias + mb);       // 32B aligned
#pragma unroll
      for (int nt = 0; nt < 4; ++nt) {
        const int n = nt * 16 + nl;
        if (n < SP) {
#pragma unroll
          for (int r = 0; r < 8; ++r)
            dst[(mb + r) * PITCH_C + n] = (_Float16)(acc[mt][nt][r] + bvv[r]);
        }
      }
    }
  }
  __syncthreads();

  // ---- P2: per-channel 7x7 scores = q[c] @ k[c] (into xT-aliased sc) ----
  for (int c = t; c < CHN; c += 256) {
    const _Float16* qr = qc + c * PITCH_C;
    const _Float16* kr = kc + c * PITCH_C;
    for (int i = 0; i < SS; ++i) {
      float qv[SS];
#pragma unroll
      for (int p = 0; p < SS; ++p) qv[p] = (float)qr[i * SS + p];
#pragma unroll
      for (int j = 0; j < SS; ++j) {
        float s = 0.f;
#pragma unroll
        for (int p = 0; p < SS; ++p) s += qv[p] * (float)kr[p * SS + j];
        sc[c * PITCH_C + i * SS + j] = (_Float16)s;
      }
    }
  }
  __syncthreads();

  // ---- P3: softmax over 512 channels at each of 49 spatial positions.
  //      4 threads per position (196 active), 128-channel chains, partials
  //      combined via LDS scratch in the dead q+k tail. ----
  const int p4 = t >> 2;       // position 0..63 (only <49 valid)
  const int g4 = t & 3;        // channel group 0..3
  const int c0 = g4 * 128, c1 = c0 + 128;
  if (p4 < SP) {
    float mx = -3.0e38f;
    for (int c = c0; c < c1; ++c)
      mx = fmaxf(mx, (float)sc[c * PITCH_C + p4]);
    redm[p4 * 4 + g4] = mx;
  }
  __syncthreads();
  if (p4 < SP) {
    const float mx = fmaxf(fmaxf(redm[p4 * 4 + 0], redm[p4 * 4 + 1]),
                           fmaxf(redm[p4 * 4 + 2], redm[p4 * 4 + 3]));
    float sum = 0.f;
    for (int c = c0; c < c1; ++c) {
      float e = __expf((float)sc[c * PITCH_C + p4] - mx);
      sum += e;
      sc[c * PITCH_C + p4] = (_Float16)e;
    }
    reds[p4 * 4 + g4] = sum;
  }
  __syncthreads();
  if (p4 < SP) {
    const float inv = 1.f / (reds[p4 * 4 + 0] + reds[p4 * 4 + 1] +
                             reds[p4 * 4 + 2] + reds[p4 * 4 + 3]);
    for (int c = c0; c < c1; ++c)
      sc[c * PITCH_C + p4] = (_Float16)((float)sc[c * PITCH_C + p4] * inv);
  }
  __syncthreads();

  // ---- P4: attn[c] = sm[c] @ v[c] -> attnT[n][c] (B layout, q+k region) ----
  for (int c = t; c < CHN; c += 256) {
    const _Float16* sr = sc + c * PITCH_C;
    const _Float16* vr = vc + c * PITCH_C;
    for (int i = 0; i < SS; ++i) {
      float sv[SS];
#pragma unroll
      for (int p = 0; p < SS; ++p) sv[p] = (float)sr[i * SS + p];
#pragma unroll
      for (int j = 0; j < SS; ++j) {
        float s = 0.f;
#pragma unroll
        for (int p = 0; p < SS; ++p) s += sv[p] * (float)vr[p * SS + j];
        aT[(i * SS + j) * PITCH_T + c] = (_Float16)s;
      }
    }
  }
  __syncthreads();

  // ---- P5: refined = Wr @ attn + br ; out = alpha*refined + x ----
  {
    const _Float16* __restrict__ W = W16 + (size_t)3 * CHN * CHN;
    const float a0 = alpha[0];
    v8f acc[4][4];
#pragma unroll
    for (int i = 0; i < 4; ++i)
#pragma unroll
      for (int j = 0; j < 4; ++j) acc[i][j] = (v8f){0, 0, 0, 0, 0, 0, 0, 0};

    for (int kk = 0; kk < CHN; kk += KSTEP) {
      v16h bf[4];
#pragma unroll
      for (int nt = 0; nt < 4; ++nt)
        bf[nt] = load_b_frag(aT + (nt * 16 + nl) * PITCH_T + kk, kb_b);
#pragma unroll
      for (int mt = 0; mt < 4; ++mt) {
        v16h af = load_a_frag(W + (size_t)(mrow + mt * 16 + nl) * CHN + kk, kb_a);
#pragma unroll
        for (int nt = 0; nt < 4; ++nt)
          acc[mt][nt] = __builtin_amdgcn_wmma_f32_16x16x32_f16(
              false, af, false, bf[nt], (short)0, acc[mt][nt], false, false);
      }
    }
    float* __restrict__ ob = out + (size_t)b * CHN * SP;
#pragma unroll
    for (int mt = 0; mt < 4; ++mt) {
      const int mb = mrow + mt * 16 + kb_a;
      v8f bvv = *(const v8f*)(br + mb);
#pragma unroll
      for (int nt = 0; nt < 4; ++nt) {
        const int n = nt * 16 + nl;
        if (n < SP) {
#pragma unroll
          for (int r = 0; r < 8; ++r) {
            const int m = mb + r;
            ob[m * SP + n] = a0 * (acc[mt][nt][r] + bvv[r]) + xb[m * SP + n];
          }
        }
      }
    }
  }
}

// ---------------------------------------------------------------------------
extern "C" void kernel_launch(void* const* d_in, const int* in_sizes, int n_in,
                              void* d_out, int out_size, void* d_ws,
                              size_t ws_size, hipStream_t stream) {
  const float* x     = (const float*)d_in[0];
  const float* Wq    = (const float*)d_in[1];
  const float* bq    = (const float*)d_in[2];
  const float* Wk    = (const float*)d_in[3];
  const float* bk    = (const float*)d_in[4];
  const float* Wv    = (const float*)d_in[5];
  const float* bv    = (const float*)d_in[6];
  const float* Wr    = (const float*)d_in[7];
  const float* br    = (const float*)d_in[8];
  const float* alpha = (const float*)d_in[9];
  float* out = (float*)d_out;

  // f16 weight pack in workspace: 4 * 512*512 * 2B = 2 MB (L2-resident)
  _Float16* W16 = (_Float16*)d_ws;
  const int nw = CHN * CHN;
  const int gw = (nw + 255) / 256;
  cvt_w_f16<<<gw, 256, 0, stream>>>(Wq, W16 + (size_t)0 * nw, nw);
  cvt_w_f16<<<gw, 256, 0, stream>>>(Wk, W16 + (size_t)1 * nw, nw);
  cvt_w_f16<<<gw, 256, 0, stream>>>(Wv, W16 + (size_t)2 * nw, nw);
  cvt_w_f16<<<gw, 256, 0, stream>>>(Wr, W16 + (size_t)3 * nw, nw);

  (void)hipFuncSetAttribute((const void*)spatial_attn_fused,
                            hipFuncAttributeMaxDynamicSharedMemorySize,
                            LDS_BYTES);
  spatial_attn_fused<<<512, 256, LDS_BYTES, stream>>>(x, W16, bq, bk, bv, br,
                                                      alpha, out);
}
